// LocalAttention_14267881357661
// MI455X (gfx1250) — matrix operand order repaired and checked by reading, not compiled, not run on hardware
//
#include <hip/hip_runtime.h>
#include <hip/hip_bf16.h>

typedef __attribute__((ext_vector_type(16))) __bf16 v16bf;
typedef __attribute__((ext_vector_type(8)))  __bf16 v8bf;
typedef __attribute__((ext_vector_type(8)))  float  v8f;

#define TOK_T 8
#define TOK_H 32
#define TOK_W 32
#define NTOK  16384   // B * N = 2 * 8192
#define HID   256
#define NHEAD 8
#define HDIM  32

// ---------------------------------------------------------------------------
// Stage 0: fp32 -> bf16 conversion (one element per thread)
// ---------------------------------------------------------------------------
__global__ void cvt_f32_bf16(const float* __restrict__ in,
                             __bf16* __restrict__ out, int n) {
  int i = blockIdx.x * blockDim.x + threadIdx.x;
  if (i < n) out[i] = (__bf16)in[i];
}

// ---------------------------------------------------------------------------
// WMMA GEMM: C[M x Nout](f32) = A[M x K](bf16) * W[Nout x K]^T(bf16)
// Each wave computes a 16 x 64 strip: one A fragment reused across 4 B tiles.
// K must be a multiple of 32; M,Nout multiples of 16; Nout multiple of 64.
// ---------------------------------------------------------------------------
__global__ void wmma_gemm_bf16(const __bf16* __restrict__ A,
                               const __bf16* __restrict__ W,
                               float* __restrict__ C,
                               int M, int Nout, int K) {
  const int wave = (blockIdx.x * blockDim.x + threadIdx.x) >> 5;
  const int lane = threadIdx.x & 31;
  const int nstrips = Nout >> 6;              // strips of 64 columns
  const int tn = wave % nstrips;
  const int tm = wave / nstrips;
  if (tm >= (M >> 4)) return;                  // wave-uniform guard
  const int m0 = tm << 4;
  const int n0 = tn << 6;

  const int mr = lane & 15;                    // A row / B column within tile
  const int kh = lane >> 4;                    // K-half select

  v8f acc0 = {}, acc1 = {}, acc2 = {}, acc3 = {};

  const __bf16* arow = A + (size_t)(m0 + mr) * K;
  const __bf16* wr0  = W + (size_t)(n0 +  0 + mr) * K;
  const __bf16* wr1  = W + (size_t)(n0 + 16 + mr) * K;
  const __bf16* wr2  = W + (size_t)(n0 + 32 + mr) * K;
  const __bf16* wr3  = W + (size_t)(n0 + 48 + mr) * K;

  #pragma unroll 4
  for (int k0 = 0; k0 < K; k0 += 32) {
    // A fragment, contiguous k order (same as B): lane group 0 holds K {0..15},
    // lane group 1 holds K {16..31} -> two contiguous 16B loads.
    v8bf alo = *(const v8bf*)(arow + k0 + kh * 16);
    v8bf ahi = *(const v8bf*)(arow + k0 + 8 + kh * 16);
    v16bf afrag;
    #pragma unroll
    for (int i = 0; i < 8; ++i) { afrag[i] = alo[i]; afrag[i + 8] = ahi[i]; }

    // B 32x16 layout: lane = column, lane group = K half -> one contiguous
    // 32B load per lane (16 consecutive K along a row of W).
    v16bf b0 = *(const v16bf*)(wr0 + k0 + kh * 16);
    v16bf b1 = *(const v16bf*)(wr1 + k0 + kh * 16);
    v16bf b2 = *(const v16bf*)(wr2 + k0 + kh * 16);
    v16bf b3 = *(const v16bf*)(wr3 + k0 + kh * 16);

    acc0 = __builtin_amdgcn_wmma_f32_16x16x32_bf16(false, afrag, false, b0,
                                                   (short)0, acc0, false, false);
    acc1 = __builtin_amdgcn_wmma_f32_16x16x32_bf16(false, afrag, false, b1,
                                                   (short)0, acc1, false, false);
    acc2 = __builtin_amdgcn_wmma_f32_16x16x32_bf16(false, afrag, false, b2,
                                                   (short)0, acc2, false, false);
    acc3 = __builtin_amdgcn_wmma_f32_16x16x32_bf16(false, afrag, false, b3,
                                                   (short)0, acc3, false, false);
  }

  // C/D layout: VGPR r holds row (r + 8*laneGroup), column = lane & 15.
  const int row0 = m0 + (kh << 3);
  const int col  = n0 + mr;
  #pragma unroll
  for (int r = 0; r < 8; ++r) {
    float* crow = C + (size_t)(row0 + r) * Nout + col;
    crow[ 0] = acc0[r];
    crow[16] = acc1[r];
    crow[32] = acc2[r];
    crow[48] = acc3[r];
  }
}

// ---------------------------------------------------------------------------
// Local 3x3x3 attention, one wave per (token, head); lane = channel d (0..31).
// Online softmax over all 27 neighbor slots (zero-padded slots contribute
// score 0 and value 0, matching the reference).
// ---------------------------------------------------------------------------
__global__ void local_attn(const float* __restrict__ qkv,
                           __bf16* __restrict__ attn_out) {
  const int wid  = (blockIdx.x * blockDim.x + threadIdx.x) >> 5;
  const int lane = threadIdx.x & 31;
  const int head = wid & (NHEAD - 1);
  const int bn   = wid >> 3;
  if (bn >= NTOK) return;

  const int b   = bn >> 13;        // / 8192
  const int n   = bn & 8191;
  const int t   = n >> 10;         // / (32*32)
  const int rem = n & 1023;
  const int hh  = rem >> 5;
  const int ww  = rem & 31;

  const float qd = qkv[(size_t)bn * 768 + head * HDIM + lane];
  const float scale = 0.17677669529663687f;   // 1/sqrt(32)

  float m = -3.402823466e38f;
  float denom = 0.f;
  float num = 0.f;

  #pragma unroll
  for (int dt = -1; dt <= 1; ++dt) {
    #pragma unroll
    for (int dh = -1; dh <= 1; ++dh) {
      #pragma unroll
      for (int dw = -1; dw <= 1; ++dw) {
        const int t2 = t + dt, h2 = hh + dh, w2 = ww + dw;
        const bool valid = ((unsigned)t2 < TOK_T) & ((unsigned)h2 < TOK_H) &
                           ((unsigned)w2 < TOK_W);
        float kd = 0.f, vd = 0.f;
        if (valid) {
          const int nb = b * 8192 + (t2 * TOK_H + h2) * TOK_W + w2;
          const float* p = qkv + (size_t)nb * 768 + head * HDIM + lane;
          kd = p[256];               // k block
          vd = p[512];               // v block
        }
        float dot = qd * kd;
        #pragma unroll
        for (int off = 16; off > 0; off >>= 1)
          dot += __shfl_xor(dot, off, 32);
        const float s    = dot * scale;
        const float mnew = fmaxf(m, s);
        const float corr = __expf(m - mnew);   // 0 on first iter (m = -inf)
        const float e    = __expf(s - mnew);
        num   = num * corr + e * vd;
        denom = denom * corr + e;
        m = mnew;
      }
    }
  }

  attn_out[(size_t)bn * HID + head * HDIM + lane] = (__bf16)(num / denom);
}

// ---------------------------------------------------------------------------
extern "C" void kernel_launch(void* const* d_in, const int* in_sizes, int n_in,
                              void* d_out, int out_size, void* d_ws, size_t ws_size,
                              hipStream_t stream) {
  const float* x     = (const float*)d_in[0];   // [2, 8192, 256]
  const float* w_qkv = (const float*)d_in[1];   // [768, 256]
  const float* w_out = (const float*)d_in[2];   // [256, 256]
  float* out = (float*)d_out;                   // [2, 8192, 256]

  char* ws = (char*)d_ws;
  __bf16* xb  = (__bf16*)ws;  ws += (size_t)NTOK * HID * 2;        // 8.39 MB
  __bf16* wqb = (__bf16*)ws;  ws += (size_t)768 * 256 * 2;         // 0.39 MB
  __bf16* wob = (__bf16*)ws;  ws += (size_t)256 * 256 * 2;         // 0.13 MB
  float*  qkv = (float*)ws;   ws += (size_t)NTOK * 768 * 4;        // 50.3 MB
  __bf16* ao  = (__bf16*)ws;                                       // 8.39 MB

  const int nx = NTOK * HID;        // 4194304
  const int nq = 768 * 256;         // 196608
  const int no = 256 * 256;         // 65536
  cvt_f32_bf16<<<(nx + 255) / 256, 256, 0, stream>>>(x, xb, nx);
  cvt_f32_bf16<<<(nq + 255) / 256, 256, 0, stream>>>(w_qkv, wqb, nq);
  cvt_f32_bf16<<<(no + 255) / 256, 256, 0, stream>>>(w_out, wob, no);

  // GEMM1: M=16384, Nout=768, K=256 -> 1024 * 12 strip-waves, 8 waves/block
  wmma_gemm_bf16<<<(1024 * 12) / 8, 256, 0, stream>>>(xb, wqb, qkv,
                                                      NTOK, 768, HID);

  // Attention: 16384 tokens * 8 heads = 131072 waves, 8 waves/block
  local_attn<<<131072 / 8, 256, 0, stream>>>(qkv, ao);

  // GEMM2: M=16384, Nout=256, K=256 -> 1024 * 4 strip-waves
  wmma_gemm_bf16<<<(1024 * 4) / 8, 256, 0, stream>>>(ao, wob, out,
                                                     NTOK, HID, HID);
}
